// LearnableShapeletWithPenalty_60636348285192
// MI455X (gfx1250) — matrix operand order repaired and checked by reading, not compiled, not run on hardware
//
#include <hip/hip_runtime.h>

// Problem constants (from reference)
#define B_  64
#define M_  16
#define T_  2048
#define L_  64
#define K_  32
#define N_  (T_ - L_ + 1)   // 1985
#define NPAD 2048           // padded n-stride: affine addressing, no clamps

#define NCHUNK 128          // n positions per block (8 waves x 16)
#define NTILE  16           // n positions per wave
#define XS_W   (NCHUNK + 64) // 192 staged values per m row
#define BPW    2            // batches per wave (amortizes pen/c3 loads)

typedef __attribute__((ext_vector_type(16))) _Float16 v16h;
typedef __attribute__((ext_vector_type(8)))  float    v8f;

// Workspace layout:
//   pen2 : [M][NPAD][16][2]  f32  (4 MB)   pen transposed+interleaved, k=(kk,kt)
//   sqs2 : [B][M][NPAD]      f32  (8 MB)   padded sliding sum of squares
//   c3   : [NPAD][16][2]     f32  (256 KB) s2[k] * sum_m pen[k,m,n]
//   xh   : [B][M][T]         f16  (4 MB)   f16-rounded x
#define PEN2_FLOATS ((size_t)M_ * NPAD * K_)
#define SQS2_FLOATS ((size_t)B_ * M_ * NPAD)
#define C3_FLOATS   ((size_t)NPAD * K_)

// ---------------------------------------------------------------------------
// Kernel 0: pen = elu(-pmap)+2, scattered into the transposed/interleaved
// layout; also init d_out to +inf bit pattern for the atomic-min reduction.
// ---------------------------------------------------------------------------
__global__ __launch_bounds__(256)
void pen_init_kernel(const float* __restrict__ pmap,
                     float* __restrict__ pen2,
                     unsigned int* __restrict__ out) {
    int i = blockIdx.x * 256 + threadIdx.x;
    const int total = K_ * M_ * N_;
    if (i < total) {
        const int n  = i % N_;
        const int km = i / N_;
        const int m  = km % M_;
        const int k  = km / M_;
        float z = -pmap[i];
        float e = (z > 0.0f) ? z : (expf(z) - 1.0f);
        pen2[((size_t)(m * NPAD + n) * 16 + (k & 15)) * 2 + (k >> 4)] = e + 2.0f;
    }
    if (i < B_ * K_) out[i] = 0x7F800000u;  // +inf
}

// ---------------------------------------------------------------------------
// Kernel 1: c3[n][kk][kt] = s2[k] * sum_m pen[k,m,n]  (b-independent term,
// hoisted out of the hot loop). s2 uses f16-rounded shapelets to match WMMA.
// ---------------------------------------------------------------------------
__global__ __launch_bounds__(256)
void c3_kernel(const float* __restrict__ shp,
               const float* __restrict__ pen2,
               float* __restrict__ c3) {
    int i = blockIdx.x * 256 + threadIdx.x;
    if (i >= N_ * K_) return;
    const int k = i & 31;
    const int n = i >> 5;
    const int kk = k & 15, kt = k >> 4;
    float s2 = 0.0f;
    #pragma unroll 8
    for (int l = 0; l < L_; ++l) {
        float sv = (float)(_Float16)shp[k * L_ + l];
        s2 += sv * sv;
    }
    float ps = 0.0f;
    for (int m = 0; m < M_; ++m)
        ps += pen2[((size_t)(m * NPAD + n) * 16 + kk) * 2 + kt];
    c3[((size_t)n * 16 + kk) * 2 + kt] = s2 * ps;
}

// ---------------------------------------------------------------------------
// Kernel 2: sqs2[b,m,n] = sum_l (f16(x[b,m,n+l]))^2, and xh = f16(x) as a
// side product (f16-rounded so x^2 - 2xs + s^2 matches the f16 WMMA term).
// ---------------------------------------------------------------------------
__global__ __launch_bounds__(256)
void sqsum_kernel(const float* __restrict__ x,
                  float* __restrict__ sqs2,
                  _Float16* __restrict__ xh) {
    __shared__ float xs[T_];
    const int row = blockIdx.x;               // b*M + m
    for (int i = threadIdx.x; i < T_; i += 256) {
        _Float16 h = (_Float16)x[row * T_ + i];
        xh[(size_t)row * T_ + i] = h;
        float v = (float)h;
        xs[i] = v * v;
    }
    __syncthreads();
    for (int n = threadIdx.x; n < N_; n += 256) {
        float s = 0.0f;
        #pragma unroll 8
        for (int l = 0; l < L_; ++l) s += xs[n + l];
        sqs2[(size_t)row * NPAD + n] = s;
    }
}

// ---------------------------------------------------------------------------
// Kernel 3: fused sliding-window GEMM (V_WMMA_F32_16X16X32_F16) + penalty
// weighting + min-reduction, 2 batches per wave.
//   cross[n,k] = sum_l x16[b,m,n+l] * s16[k,l]
//   wd[n,k]    = sum_m pen[k,m,n]*(sqsum - 2*cross) + c3[n,k]
//   out[b,k]   = atomic min over n
// ---------------------------------------------------------------------------
__global__ __launch_bounds__(256)
void shapelet_wmma_kernel(const _Float16* __restrict__ xh,
                          const float* __restrict__ shp,
                          const float* __restrict__ pen2,
                          const float* __restrict__ sqs2,
                          const float* __restrict__ c3,
                          unsigned int* __restrict__ out) {
    __shared__ _Float16 xs[BPW][M_][XS_W];    // 2 x 16 x 192 f16 = 12 KB

    const int b0    = blockIdx.y * BPW;
    const int nbase = blockIdx.x * NCHUNK;
    const int tid   = threadIdx.x;

    // Stage pre-converted f16 x as 32-bit copies (pair granularity, clamped
    // at the row end; garbage edge windows are masked with +inf at the min).
    {
        const unsigned int* src = (const unsigned int*)xh;
        unsigned int* dst = (unsigned int*)xs;
        const int PAIRS = XS_W / 2;           // 96 dwords per row
        for (int i = tid; i < BPW * M_ * PAIRS; i += 256) {
            const int bb = i / (M_ * PAIRS);
            const int r  = i % (M_ * PAIRS);
            const int m  = r / PAIRS, j = r % PAIRS;
            int idx = nbase / 2 + j;
            if (idx > T_ / 2 - 1) idx = T_ / 2 - 1;
            dst[i] = src[((size_t)(b0 + bb) * M_ + m) * (T_ / 2) + idx];
        }
    }
    __syncthreads();

    const int wave   = tid >> 5;
    const int lane   = tid & 31;
    const int n0     = wave * NTILE;
    const int nglob0 = nbase + n0;
    if (nglob0 >= N_) return;                 // whole wave out of range

    const int hi   = (lane >= 16) ? 1 : 0;
    const int lcol = lane & 15;               // A row / D column index
    const int nrow = nglob0 + hi * 8;         // first n this half-wave holds

    // --- B fragments (k-tile x l-chunk): dense 32x16 B layout: lanes 0-15
    // hold K(=l)=0..7,16..23, lanes 16-31 hold K=8..15,24..31 (matches A).
    v16h bf[2][2];
    for (int kt = 0; kt < 2; ++kt) {
        const int k = kt * 16 + lcol;
        for (int lc = 0; lc < 2; ++lc) {
            const int lb = lc * 32 + hi * 8;
            #pragma unroll
            for (int j = 0; j < 16; ++j)
                bf[kt][lc][j] = (_Float16)shp[k * L_ + lb + j + (j >= 8 ? 8 : 0)];
        }
    }

    // Affine epilogue base pointers (all loop offsets become immediates)
    const float2* __restrict__ p2  =
        (const float2*)pen2 + ((size_t)nrow * 16 + lcol);       // + m*32768 + j*16
    const float2* __restrict__ c3b =
        (const float2*)c3 + ((size_t)nrow * 16 + lcol);         // + j*16
    const float4* __restrict__ sq4[BPW];
    #pragma unroll
    for (int bb = 0; bb < BPW; ++bb)
        sq4[bb] = (const float4*)(sqs2 + ((size_t)(b0 + bb) * M_ * NPAD + nrow));

    float wd[BPW][2][8];
    #pragma unroll
    for (int bb = 0; bb < BPW; ++bb)
        #pragma unroll
        for (int kt = 0; kt < 2; ++kt)
            #pragma unroll
            for (int j = 0; j < 8; ++j) wd[bb][kt][j] = 0.0f;

    for (int m = 0; m < M_; ++m) {
        // Shared (b-invariant) penalty weights for this m: 8 coalesced b64.
        float2 pp[8];
        #pragma unroll
        for (int j = 0; j < 8; ++j)
            pp[j] = p2[(size_t)m * (NPAD * 16) + j * 16];

        #pragma unroll
        for (int bb = 0; bb < BPW; ++bb) {
            // A fragments per l-chunk, interleaved 16-bit 16x32 A layout.
            v16h af[2];
            #pragma unroll
            for (int lc = 0; lc < 2; ++lc) {
                #pragma unroll
                for (int v = 0; v < 8; ++v) {
                    const int ks = 2 * v + (v < 4 ? (hi ? 8 : 0) : (hi ? 16 : 8));
                    const int base = n0 + lcol + lc * 32 + ks;   // row = lcol
                    af[lc][2 * v]     = xs[bb][m][base];
                    af[lc][2 * v + 1] = xs[bb][m][base + 1];
                }
            }

            v8f cacc[2];
            cacc[0] = (v8f){};
            cacc[1] = (v8f){};
            cacc[0] = __builtin_amdgcn_wmma_f32_16x16x32_f16(false, af[0], false, bf[0][0], (short)0, cacc[0], false, false);
            cacc[0] = __builtin_amdgcn_wmma_f32_16x16x32_f16(false, af[1], false, bf[0][1], (short)0, cacc[0], false, false);
            cacc[1] = __builtin_amdgcn_wmma_f32_16x16x32_f16(false, af[0], false, bf[1][0], (short)0, cacc[1], false, false);
            cacc[1] = __builtin_amdgcn_wmma_f32_16x16x32_f16(false, af[1], false, bf[1][1], (short)0, cacc[1], false, false);

            const float4 sqa = sq4[bb][(size_t)m * (NPAD / 4)];
            const float4 sqb = sq4[bb][(size_t)m * (NPAD / 4) + 1];
            const float sqv[8] = { sqa.x, sqa.y, sqa.z, sqa.w,
                                   sqb.x, sqb.y, sqb.z, sqb.w };
            #pragma unroll
            for (int j = 0; j < 8; ++j) {
                const float t0 = sqv[j] - 2.0f * cacc[0][j];  // one pk_fma (neg)
                const float t1 = sqv[j] - 2.0f * cacc[1][j];
                wd[bb][0][j] = fmaf(pp[j].x, t0, wd[bb][0][j]);
                wd[bb][1][j] = fmaf(pp[j].y, t1, wd[bb][1][j]);
            }
        }
    }

    // Add hoisted c3 term (shared across bb), min over n, reduce, atomic.
    // wd >= 0 (pen > 1, raw >= 0) -> uint bit-pattern order == float order.
    float2 cc[8];
    #pragma unroll
    for (int j = 0; j < 8; ++j) cc[j] = c3b[j * 16];

    #pragma unroll
    for (int bb = 0; bb < BPW; ++bb) {
        float mn0 = __builtin_inff(), mn1 = __builtin_inff();
        #pragma unroll
        for (int j = 0; j < 8; ++j) {
            const int n = nrow + j;
            const float v0 = (n < N_) ? (wd[bb][0][j] + cc[j].x) : __builtin_inff();
            const float v1 = (n < N_) ? (wd[bb][1][j] + cc[j].y) : __builtin_inff();
            mn0 = fminf(mn0, v0);
            mn1 = fminf(mn1, v1);
        }
        mn0 = fminf(mn0, __shfl_xor(mn0, 16, 32));
        mn1 = fminf(mn1, __shfl_xor(mn1, 16, 32));
        if (lane < 16) {
            atomicMin(&out[(b0 + bb) * K_ + lane],      __float_as_uint(mn0));
            atomicMin(&out[(b0 + bb) * K_ + 16 + lane], __float_as_uint(mn1));
        }
    }
}

// ---------------------------------------------------------------------------
extern "C" void kernel_launch(void* const* d_in, const int* in_sizes, int n_in,
                              void* d_out, int out_size, void* d_ws, size_t ws_size,
                              hipStream_t stream) {
    const float* x    = (const float*)d_in[0];   // (B, M, T)
    const float* shp  = (const float*)d_in[1];   // (K, L)
    const float* pmap = (const float*)d_in[2];   // (K, M, N)

    float* pen2 = (float*)d_ws;
    float* sqs2 = pen2 + PEN2_FLOATS;
    float* c3   = sqs2 + SQS2_FLOATS;
    _Float16* xh = (_Float16*)(c3 + C3_FLOATS);
    unsigned int* out = (unsigned int*)d_out;    // (B, K)

    {
        const int total  = K_ * M_ * N_;
        pen_init_kernel<<<(total + 255) / 256, 256, 0, stream>>>(pmap, pen2, out);
    }
    c3_kernel<<<(N_ * K_ + 255) / 256, 256, 0, stream>>>(shp, pen2, c3);
    sqsum_kernel<<<B_ * M_, 256, 0, stream>>>(x, sqs2, xh);

    dim3 grid((N_ + NCHUNK - 1) / NCHUNK, B_ / BPW);   // (16, 32)
    shapelet_wmma_kernel<<<grid, 256, 0, stream>>>(xh, shp, pen2, sqs2, c3, out);
}
